// RNN_Classifier_171798692355
// MI455X (gfx1250) — matrix operand order repaired and checked by reading, not compiled, not run on hardware
//
#include <hip/hip_runtime.h>

typedef __attribute__((ext_vector_type(16))) __bf16 v16bf;
typedef __attribute__((ext_vector_type(8)))  __bf16 v8bf;
typedef __attribute__((ext_vector_type(4)))  __bf16 v4bf;
typedef __attribute__((ext_vector_type(8)))  float  v8f;
typedef __attribute__((ext_vector_type(4)))  float  v4f;

#define SEQ     128
#define BATCH   8192
#define INF     28
#define HID     128
#define CLASSES 10

__device__ __forceinline__ v16bf join8(v8bf lo, v8bf hi) {
    return __builtin_shufflevector(lo, hi, 0,1,2,3,4,5,6,7,8,9,10,11,12,13,14,15);
}

// One block = 16 batch rows, 8 waves; wave w owns hidden columns [16w,16w+16).
// h resident in LDS (bf16) across all SEQ steps; weight fragments resident in VGPRs.
// x tile double-buffered in LDS: step t+1 is fetched (b128) INSIDE the compute
// phase (after the first barrier) so its latency hides under the WMMAs instead
// of being drained by the barrier fence.
__global__ __launch_bounds__(256)
void rnn_fused_kernel(const float* __restrict__ x,
                      const float* __restrict__ W_ih,
                      const float* __restrict__ W_hh,
                      const float* __restrict__ b_ih,
                      const float* __restrict__ b_hh,
                      const float* __restrict__ W_dense,
                      const float* __restrict__ b_dense,
                      float* __restrict__ out)
{
    __shared__ __bf16 hbuf[16 * HID];      // [m][k] current hidden-state tile
    __shared__ __bf16 xbuf[2][16 * 32];    // [m][i] x tiles, K padded 28 -> 32

    const int tid  = threadIdx.x;
    const int lane = tid & 31;
    const int wave = tid >> 5;
    const int half = lane >> 4;    // lane group 0-15 vs 16-31
    const int mrow = lane & 15;    // A-matrix row for this lane
    const int n    = lane & 15;    // C/D column for this lane
    const int j0   = wave * 16;    // hidden-column tile of this wave
    const int b0   = blockIdx.x * 16;

    // ---- x staging: 112 threads each own one float4 of the 16x28 tile ----
    const bool loader = (tid < (16 * INF) / 4);       // tid < 112
    const int  lm = (tid * 4) / INF;                  // tile row
    const int  li = (tid * 4) % INF;                  // col (multiple of 4)
    const size_t stepstride = (size_t)BATCH * INF;    // elements per timestep
    const float* xsrc = x + (size_t)(b0 + lm) * INF + li;

    // ---- loop-invariant B fragments, gathered once (weights live in L2) ----
    // B layout (16-bit 32x16 KxN): lane = col n (+16 lanes for K-half), elem e -> K = 8*half + e (e<8), 16 + 8*half + (e-8) (e>=8)
    v16bf Bih;
    #pragma unroll
    for (int e = 0; e < 16; ++e) {
        int k = 8 * half + e + (e >= 8 ? 8 : 0);
        float wv = (k < INF) ? W_ih[(j0 + n) * INF + k] : 0.f;
        Bih[e] = (__bf16)wv;
    }
    v16bf Bhh[4];
    #pragma unroll
    for (int kt = 0; kt < 4; ++kt) {
        #pragma unroll
        for (int e = 0; e < 16; ++e) {
            int k = kt * 32 + 8 * half + e + (e >= 8 ? 8 : 0);
            Bhh[kt][e] = (__bf16)W_hh[(j0 + n) * HID + k];
        }
    }
    const float biasv = b_ih[j0 + n] + b_hh[j0 + n];

    // zero initial h; zero the static pad columns (i=28..31) of both x buffers once
    for (int i2 = tid; i2 < 16 * HID; i2 += 256) hbuf[i2] = (__bf16)0.f;
    if (tid < 32) {
        int buf = tid >> 4, m = tid & 15;
        #pragma unroll
        for (int p = 0; p < 4; ++p) xbuf[buf][m * 32 + INF + p] = (__bf16)0.f;
    }

    // pre-stage t = 0 into buffer 0
    if (loader) {
        v4f r = *(const v4f*)xsrc;
        v4bf o;
        #pragma unroll
        for (int p = 0; p < 4; ++p) o[p] = (__bf16)r[p];
        *(v4bf*)&xbuf[0][lm * 32 + li] = o;
    }

    int cur = 0;
    for (int t = 0; t < SEQ; ++t) {
        const int nxt = cur ^ 1;

        __syncthreads();   // xbuf[cur] ready; prev-step hbuf writes visible

        // issue next step's global fetch AFTER the barrier so the barrier fence
        // doesn't drain it; it is consumed only after the WMMAs below.
        v4f r;
        if (loader) {
            int tn = (t + 1 < SEQ) ? (t + 1) : t;   // clamp (last store never read)
            r = *(const v4f*)(xsrc + (size_t)tn * stepstride);
        }

        v8f acc;
        #pragma unroll
        for (int v = 0; v < 8; ++v) acc[v] = biasv;   // bias folded into accumulator

        // input projection: one WMMA, K padded to 32
        // A-frag (16-bit 16x32): lane holds K = {8*half..+8} and {16+8*half..+8}
        {
            const v8bf* p = (const v8bf*)&xbuf[cur][mrow * 32 + 8 * half];
            v16bf a = join8(p[0], p[2]);
            acc = __builtin_amdgcn_wmma_f32_16x16x32_bf16(
                      false, a, false, Bih, (short)0, acc, false, false);
        }
        // recurrence: K = 128 over 4 WMMAs
        #pragma unroll
        for (int kt = 0; kt < 4; ++kt) {
            const v8bf* p = (const v8bf*)&hbuf[mrow * HID + kt * 32 + 8 * half];
            v16bf a = join8(p[0], p[2]);
            acc = __builtin_amdgcn_wmma_f32_16x16x32_bf16(
                      false, a, false, Bhh[kt], (short)0, acc, false, false);
        }

        // pack next x tile into the other buffer while WMMAs drain
        if (loader) {
            v4bf o;
            #pragma unroll
            for (int p = 0; p < 4; ++p) o[p] = (__bf16)r[p];
            *(v4bf*)&xbuf[nxt][lm * 32 + li] = o;
        }

        __syncthreads();   // all waves finished reading hbuf / xbuf[cur]

        // relu, write new h tile (C/D layout: VGPR v -> row v + 8*half)
        #pragma unroll
        for (int v = 0; v < 8; ++v) {
            float hv = acc[v] > 0.f ? acc[v] : 0.f;
            hbuf[(v + 8 * half) * HID + j0 + n] = (__bf16)hv;
        }
        cur = nxt;
    }
    __syncthreads();

    // dense head: out = h @ W_dense^T + b_dense, classes padded 10 -> 16.
    // Wave-uniform branch -> EXEC all ones inside -> WMMA legal.
    if (wave == 0) {
        v8f d;
        float bd = (n < CLASSES) ? b_dense[n] : 0.f;
        #pragma unroll
        for (int v = 0; v < 8; ++v) d[v] = bd;
        #pragma unroll
        for (int kt = 0; kt < 4; ++kt) {
            v16bf bfrag;
            #pragma unroll
            for (int e = 0; e < 16; ++e) {
                int k = kt * 32 + 8 * half + e + (e >= 8 ? 8 : 0);
                float wv = (n < CLASSES) ? W_dense[n * HID + k] : 0.f;
                bfrag[e] = (__bf16)wv;
            }
            const v8bf* p = (const v8bf*)&hbuf[mrow * HID + kt * 32 + 8 * half];
            v16bf a = join8(p[0], p[2]);
            d = __builtin_amdgcn_wmma_f32_16x16x32_bf16(
                    false, a, false, bfrag, (short)0, d, false, false);
        }
        if (n < CLASSES) {
            #pragma unroll
            for (int v = 0; v < 8; ++v)
                out[(size_t)(b0 + v + 8 * half) * CLASSES + n] = d[v];
        }
    }
}

extern "C" void kernel_launch(void* const* d_in, const int* in_sizes, int n_in,
                              void* d_out, int out_size, void* d_ws, size_t ws_size,
                              hipStream_t stream) {
    const float* x       = (const float*)d_in[0];
    const float* W_ih    = (const float*)d_in[1];
    const float* W_hh    = (const float*)d_in[2];
    const float* b_ih    = (const float*)d_in[3];
    const float* b_hh    = (const float*)d_in[4];
    const float* W_dense = (const float*)d_in[5];
    const float* b_dense = (const float*)d_in[6];
    float* out = (float*)d_out;

    dim3 grid(BATCH / 16);
    dim3 block(256);
    hipLaunchKernelGGL(rnn_fused_kernel, grid, block, 0, stream,
                       x, W_ih, W_hh, b_ih, b_hh, W_dense, b_dense, out);
}
